// QLSTM_65481071409306
// MI455X (gfx1250) — matrix operand order repaired and checked by reading, not compiled, not run on hardware
//
#include <hip/hip_runtime.h>
#include <hip/hip_bf16.h>
#include <math.h>

typedef __attribute__((ext_vector_type(16))) _Float16 v16h;
typedef __attribute__((ext_vector_type(8)))  float    v8f;

#define NQ     8
#define QDEPTH 2
#define TT     512
#define BB     256
#define DD     256
#define HH     256
#define ROWS   16   // batch rows per block (B=256 -> 16 blocks)

// ---------------------------------------------------------------------------
// Pack Wf/Wi/Wg/Wo ([8][512] f32) into Wc [32][512] f16 and biasc[32] f32.
// ---------------------------------------------------------------------------
__global__ void qlstm_prep(const float* __restrict__ Wf, const float* __restrict__ Wi,
                           const float* __restrict__ Wg, const float* __restrict__ Wo,
                           const float* __restrict__ bf, const float* __restrict__ bi,
                           const float* __restrict__ bg, const float* __restrict__ bo,
                           _Float16* __restrict__ Wc, float* __restrict__ biasc) {
  const int tid = threadIdx.x;
  const float* Ws[4] = {Wf, Wi, Wg, Wo};
  const float* bs[4] = {bf, bi, bg, bo};
  for (int idx = tid; idx < 32 * 512; idx += 256) {
    int row = idx >> 9, k = idx & 511;
    int g = row >> 3, q = row & 7;
    Wc[idx] = (_Float16)Ws[g][q * 512 + k];
  }
  if (tid < 32) biasc[tid] = bs[tid >> 3][tid & 7];
}

// ---------------------------------------------------------------------------
// Persistent scan kernel: each block owns 16 batch rows for all 512 steps.
// Phase A: angles = [x_t | h] @ Wc^T + bias  via v_wmma_f32_16x16x32_f16,
//          K split across 4 waves; x-half uses global_load_b128, h-half ds_load.
// Phase B: 8-qubit statevector sim, one wave per (gate,sample), shfl_xor gates;
//          also prefetches x[t+1] (global_prefetch_b8).
// Phase C: act(z) @ Wp^T + bp, LSTM cell update, h -> LDS + global out
// ---------------------------------------------------------------------------
__global__ void __launch_bounds__(256)
qlstm_scan(const float* __restrict__ x,        // [T,B,D]
           const _Float16* __restrict__ Wc,    // [32][512] packed gate weights
           const float* __restrict__ biasc,    // [32]
           const float* __restrict__ rysf, const float* __restrict__ rysi,
           const float* __restrict__ rysg, const float* __restrict__ ryso,
           const float* __restrict__ Wp,       // [256][8]
           const float* __restrict__ bp,       // [256]
           float* __restrict__ out)            // outs[T,B,H] ++ hx[B,H] ++ cx[B,H]
{
  __shared__ _Float16 sWc[32][516];     // padded rows (1032 B stride)
  __shared__ float    sWp[256][8];
  __shared__ float    sbp[256];
  __shared__ float    sh[ROWS][257];    // recurrent h (padded)
  __shared__ float    sang[ROWS][33];   // x-half partial + bias
  __shared__ float    sang2[ROWS][33];  // h-half partial
  __shared__ float    sz[4][ROWS][NQ];  // <Z> then activated values
  __shared__ float    sryc[4][QDEPTH][NQ];
  __shared__ float    srys[4][QDEPTH][NQ];

  const int tid  = threadIdx.x;
  const int wave = tid >> 5;
  const int lane = tid & 31;
  const int b0   = blockIdx.x * ROWS;

  // ---- one-time LDS fills ----
  for (int i = tid; i < 32 * 512; i += 256) sWc[i >> 9][i & 511] = Wc[i];
  for (int i = tid; i < 256 * 8; i += 256)  sWp[i >> 3][i & 7]   = Wp[i];
  sbp[tid] = bp[tid];
  for (int i = tid; i < ROWS * HH; i += 256) sh[i >> 8][i & 255] = 0.0f;
  if (tid < 4 * QDEPTH * NQ) {
    int g = tid >> 4, d = (tid >> 3) & 1, q = tid & 7;
    const float* rp = (g == 0) ? rysf : (g == 1) ? rysi : (g == 2) ? rysg : ryso;
    float th = 0.5f * rp[d * NQ + q];
    sryc[g][d][q] = cosf(th);
    srys[g][d][q] = sinf(th);
  }
  float creg[ROWS];
  #pragma unroll
  for (int s = 0; s < ROWS; ++s) creg[s] = 0.0f;

  const int mrow = lane & 15;   // A-row / B-col within tile
  const int hi   = lane >> 4;   // lane half selects K-chunk

  for (int t = 0; t < TT; ++t) {
    __syncthreads();  // sh from previous step is ready

    // ====== Phase A: gate pre-activations, K=512 split over 4 waves ======
    if (wave < 4) {
      const int nt  = wave & 1;         // N-tile (gate-cols 0-15 / 16-31)
      const int col = nt * 16 + mrow;
      v8f acc = {};
      if (wave < 2) {
        // ---- x-half: K = 0..255, A from global memory ----
        const float* xrow = x + ((size_t)t * BB + (b0 + mrow)) * DD;
        #pragma unroll
        for (int kk8 = 0; kk8 < 8; ++kk8) {
          const float* arow = xrow + kk8 * 32 + hi * 16;
          v16h A;
          #pragma unroll
          for (int j = 0; j < 8; ++j) A[j]     = (_Float16)arow[j];
          #pragma unroll
          for (int j = 0; j < 8; ++j) A[j + 8] = (_Float16)arow[8 + j];
          const _Float16* brow = &sWc[col][kk8 * 32 + hi * 16];
          v16h Bf;
          #pragma unroll
          for (int j = 0; j < 16; ++j) Bf[j] = brow[j];
          acc = __builtin_amdgcn_wmma_f32_16x16x32_f16(
                    false, A, false, Bf, (short)0, acc, false, false);
        }
        const float bv = biasc[col];
        #pragma unroll
        for (int r = 0; r < 8; ++r) sang[r + 8 * hi][col] = acc[r] + bv;
      } else {
        // ---- h-half: K = 256..511, A from LDS ----
        #pragma unroll
        for (int kk8 = 0; kk8 < 8; ++kk8) {
          const float* arow = &sh[mrow][kk8 * 32 + hi * 16];
          v16h A;
          #pragma unroll
          for (int j = 0; j < 8; ++j) A[j]     = (_Float16)arow[j];
          #pragma unroll
          for (int j = 0; j < 8; ++j) A[j + 8] = (_Float16)arow[8 + j];
          const _Float16* brow = &sWc[col][(8 + kk8) * 32 + hi * 16];
          v16h Bf;
          #pragma unroll
          for (int j = 0; j < 16; ++j) Bf[j] = brow[j];
          acc = __builtin_amdgcn_wmma_f32_16x16x32_f16(
                    false, A, false, Bf, (short)0, acc, false, false);
        }
        #pragma unroll
        for (int r = 0; r < 8; ++r) sang2[r + 8 * hi][col] = acc[r];
      }
    }
    // prefetch next timestep's x tile while qsim runs (global_prefetch_b8)
    if (t + 1 < TT) {
      const float* nx = x + ((size_t)(t + 1) * BB + (b0 + (tid >> 4))) * DD
                          + (tid & 15) * 16;
      __builtin_prefetch(nx, 0, 3);
    }
    __syncthreads();

    // ================= Phase B: quantum sims (one wave per sim) ==========
    // amp index a(8b): bits 7..3 = qubits 0..4 -> lane; bits 2..0 = qubits 5..7
    for (int round = 0; round < 8; ++round) {
      const int sim = wave * 8 + round;   // 64 sims = 4 gates x 16 samples
      const int g = sim >> 4;
      const int s = sim & 15;
      float ar[8], ai[8];
      #pragma unroll
      for (int j = 0; j < 8; ++j) { ar[j] = 0.0f; ai[j] = 0.0f; }
      if (lane == 0) ar[0] = 1.0f;

      // ---- RX(angle[q]) on each qubit ----
      #pragma unroll
      for (int q = 0; q < NQ; ++q) {
        const int gc = g * 8 + q;
        float th = 0.5f * (sang[s][gc] + sang2[s][gc]);
        float sn, c;
        __sincosf(th, &sn, &c);
        if (q >= 5) {                       // lane-local pair mix
          const int m = 1 << (7 - q);
          #pragma unroll
          for (int j = 0; j < 8; ++j) {
            if ((j & m) == 0) {
              const int j2 = j | m;
              float n0r = c * ar[j]  + sn * ai[j2];
              float n0i = c * ai[j]  - sn * ar[j2];
              float n1r = c * ar[j2] + sn * ai[j];
              float n1i = c * ai[j2] - sn * ar[j];
              ar[j] = n0r; ai[j] = n0i; ar[j2] = n1r; ai[j2] = n1i;
            }
          }
        } else {                            // cross-lane butterfly
          const int lm = 1 << (4 - q);
          #pragma unroll
          for (int j = 0; j < 8; ++j) {
            float pr = __shfl_xor(ar[j], lm, 32);
            float pi = __shfl_xor(ai[j], lm, 32);
            float nr = c * ar[j] + sn * pi;   // symmetric for both halves
            float ni = c * ai[j] - sn * pr;
            ar[j] = nr; ai[j] = ni;
          }
        }
      }

      // ---- DEPTH x (RY layer + CNOT chain) ----
      #pragma unroll
      for (int d = 0; d < QDEPTH; ++d) {
        #pragma unroll
        for (int q = 0; q < NQ; ++q) {
          const float c = sryc[g][d][q], sn = srys[g][d][q];
          if (q >= 5) {
            const int m = 1 << (7 - q);
            #pragma unroll
            for (int j = 0; j < 8; ++j) {
              if ((j & m) == 0) {
                const int j2 = j | m;
                float n0r = c * ar[j] - sn * ar[j2];
                float n0i = c * ai[j] - sn * ai[j2];
                float n1r = sn * ar[j] + c * ar[j2];
                float n1i = sn * ai[j] + c * ai[j2];
                ar[j] = n0r; ai[j] = n0i; ar[j2] = n1r; ai[j2] = n1i;
              }
            }
          } else {
            const int lm = 1 << (4 - q);
            const float sgn = (lane & lm) ? sn : -sn;
            #pragma unroll
            for (int j = 0; j < 8; ++j) {
              float pr = __shfl_xor(ar[j], lm, 32);
              float pi = __shfl_xor(ai[j], lm, 32);
              ar[j] = c * ar[j] + sgn * pr;
              ai[j] = c * ai[j] + sgn * pi;
            }
          }
        }
        // CNOT(q -> q+1), q = 0..3: both bits in lane index
        #pragma unroll
        for (int q = 0; q < 4; ++q) {
          const int lmT = 1 << (3 - q);
          const int cm  = 1 << (4 - q);
          const bool sw = (lane & cm) != 0;
          #pragma unroll
          for (int j = 0; j < 8; ++j) {
            float pr = __shfl_xor(ar[j], lmT, 32);
            float pi = __shfl_xor(ai[j], lmT, 32);
            if (sw) { ar[j] = pr; ai[j] = pi; }
          }
        }
        // CNOT(4 -> 5): control lane bit0, target local bit2
        if (lane & 1) {
          #pragma unroll
          for (int j = 0; j < 4; ++j) {
            float tr = ar[j]; ar[j] = ar[j + 4]; ar[j + 4] = tr;
            float ti = ai[j]; ai[j] = ai[j + 4]; ai[j + 4] = ti;
          }
        }
        // CNOT(5 -> 6): local, swap (4,6),(5,7)
        { float tr;
          tr = ar[4]; ar[4] = ar[6]; ar[6] = tr;  tr = ai[4]; ai[4] = ai[6]; ai[6] = tr;
          tr = ar[5]; ar[5] = ar[7]; ar[7] = tr;  tr = ai[5]; ai[5] = ai[7]; ai[7] = tr; }
        // CNOT(6 -> 7): local, swap (2,3),(6,7)
        { float tr;
          tr = ar[2]; ar[2] = ar[3]; ar[3] = tr;  tr = ai[2]; ai[2] = ai[3]; ai[3] = tr;
          tr = ar[6]; ar[6] = ar[7]; ar[7] = tr;  tr = ai[6]; ai[6] = ai[7]; ai[7] = tr; }
      }

      // ---- measurement: all 8 <Z_q> in one pass ----
      float p[8], psum = 0.0f;
      #pragma unroll
      for (int j = 0; j < 8; ++j) { p[j] = ar[j] * ar[j] + ai[j] * ai[j]; psum += p[j]; }
      float zq[8];
      zq[5] = (p[0] + p[1] + p[2] + p[3]) - (p[4] + p[5] + p[6] + p[7]);
      zq[6] = (p[0] + p[1] + p[4] + p[5]) - (p[2] + p[3] + p[6] + p[7]);
      zq[7] = (p[0] + p[2] + p[4] + p[6]) - (p[1] + p[3] + p[5] + p[7]);
      #pragma unroll
      for (int q = 0; q < 5; ++q)
        zq[q] = ((lane >> (4 - q)) & 1) ? -psum : psum;
      #pragma unroll
      for (int q = 0; q < 8; ++q) {
        #pragma unroll
        for (int off = 16; off > 0; off >>= 1)
          zq[q] += __shfl_xor(zq[q], off, 32);
      }
      if (lane == 0) {
        #pragma unroll
        for (int q = 0; q < 8; ++q) sz[g][s][q] = zq[q];
      }
    }
    __syncthreads();

    // ---- activations in place: sigmoid(f,i,o), tanh(g) ----
    #pragma unroll
    for (int it = 0; it < 2; ++it) {
      int idx = tid + it * 256;          // 0..511 = 4*16*8
      int g = idx >> 7, s = (idx >> 3) & 15, q = idx & 7;
      float v = sz[g][s][q];
      sz[g][s][q] = (g == 2) ? tanhf(v) : (1.0f / (1.0f + __expf(-v)));
    }
    __syncthreads();

    // ================= Phase C: projection + LSTM cell update ============
    const int col = tid;
    float* outT = out + (size_t)t * (BB * HH);
    #pragma unroll 4
    for (int s = 0; s < ROWS; ++s) {
      float fv = sbp[col], iv = fv, gv = fv, ov = fv;
      #pragma unroll
      for (int q = 0; q < 8; ++q) {
        const float w = sWp[col][q];
        fv += sz[0][s][q] * w;
        iv += sz[1][s][q] * w;
        gv += sz[2][s][q] * w;
        ov += sz[3][s][q] * w;
      }
      const float cn = fv * creg[s] + iv * gv;
      creg[s] = cn;
      const float hv = ov * tanhf(cn);
      sh[s][col] = hv;
      outT[(size_t)(b0 + s) * HH + col] = hv;
    }
  }

  // ---- final carries hx, cx ----
  float* hx = out + (size_t)TT * BB * HH;
  float* cx = hx + (size_t)BB * HH;
  for (int s = 0; s < ROWS; ++s) {
    hx[(size_t)(b0 + s) * HH + tid] = sh[s][tid];
    cx[(size_t)(b0 + s) * HH + tid] = creg[s];
  }
}

// ---------------------------------------------------------------------------
extern "C" void kernel_launch(void* const* d_in, const int* in_sizes, int n_in,
                              void* d_out, int out_size, void* d_ws, size_t ws_size,
                              hipStream_t stream) {
  (void)in_sizes; (void)n_in; (void)out_size; (void)ws_size;
  const float* x    = (const float*)d_in[0];
  const float* Wf   = (const float*)d_in[1];
  const float* bf   = (const float*)d_in[2];
  const float* Wi   = (const float*)d_in[3];
  const float* bi   = (const float*)d_in[4];
  const float* Wg   = (const float*)d_in[5];
  const float* bg   = (const float*)d_in[6];
  const float* Wo   = (const float*)d_in[7];
  const float* bo   = (const float*)d_in[8];
  const float* rysf = (const float*)d_in[9];
  const float* rysi = (const float*)d_in[10];
  const float* rysg = (const float*)d_in[11];
  const float* ryso = (const float*)d_in[12];
  const float* Wp   = (const float*)d_in[13];
  const float* bp   = (const float*)d_in[14];

  _Float16* Wc    = (_Float16*)d_ws;                       // 32*512*2 = 32 KB
  float*    biasc = (float*)((char*)d_ws + 32 * 512 * 2);  // 128 B

  hipLaunchKernelGGL(qlstm_prep, dim3(1), dim3(256), 0, stream,
                     Wf, Wi, Wg, Wo, bf, bi, bg, bo, Wc, biasc);
  hipLaunchKernelGGL(qlstm_scan, dim3(BB / ROWS), dim3(256), 0, stream,
                     x, Wc, biasc, rysf, rysi, rysg, ryso, Wp, bp, (float*)d_out);
}
